// VisionTransformer_8924942041933
// MI455X (gfx1250) — matrix operand order repaired and checked by reading, not compiled, not run on hardware
//
#include <hip/hip_runtime.h>
#include <hip/hip_bf16.h>
#include <math.h>

// ---------------------------------------------------------------------------
// Types for CDNA5 WMMA (wave32): v_wmma_f32_16x16x32_bf16
// ---------------------------------------------------------------------------
typedef __bf16 bhalf8  __attribute__((ext_vector_type(8)));
typedef __bf16 bhalf16 __attribute__((ext_vector_type(16)));
typedef float  float8  __attribute__((ext_vector_type(8)));
typedef unsigned int uint4v __attribute__((ext_vector_type(4)));
typedef int int4v __attribute__((ext_vector_type(4)));
typedef int int8v __attribute__((ext_vector_type(8)));

__device__ __forceinline__ bhalf16 join8(bhalf8 lo, bhalf8 hi) {
  return __builtin_shufflevector(lo, hi, 0,1,2,3,4,5,6,7,8,9,10,11,12,13,14,15);
}
__device__ __forceinline__ float8 zero8() {
  float8 z = {0.f,0.f,0.f,0.f,0.f,0.f,0.f,0.f};
  return z;
}

// TDM (Tensor Data Mover) availability: guarded so both toolchains compile.
#if defined(__HIP_DEVICE_COMPILE__) && defined(__has_builtin)
#if __has_builtin(__builtin_amdgcn_tensor_load_to_lds) && __has_builtin(__builtin_amdgcn_s_wait_tensorcnt)
#define GEMM_TDM 1
#endif
#endif
#ifndef GEMM_TDM
#define GEMM_TDM 0
#endif

// ---------------------------------------------------------------------------
// Problem constants
// ---------------------------------------------------------------------------
#define VIT_B      32
#define VIT_NTOK   384          // 2*64 + 256
#define VIT_DIM    768
#define VIT_MLP    3072
#define VIT_HEADS  12
#define VIT_HD     64
#define VIT_DEPTH  12
#define VIT_M      (VIT_B * VIT_NTOK)   // 12288 rows of activations

// ---------------------------------------------------------------------------
// fp32 -> bf16 weight conversion (grid-strided)
// ---------------------------------------------------------------------------
__global__ void cvt_f32_bf16(const float* __restrict__ in, __bf16* __restrict__ out, long n) {
  long i = (long)blockIdx.x * blockDim.x + threadIdx.x;
  long stride = (long)gridDim.x * blockDim.x;
  for (; i < n; i += stride) out[i] = (__bf16)in[i];
}

// ---------------------------------------------------------------------------
// LayerNorm: fp32 x[row,768] -> bf16 y[row,768]; one block per token row
// ---------------------------------------------------------------------------
__global__ __launch_bounds__(256) void layernorm_bf16(
    const float* __restrict__ X, const float* __restrict__ g,
    const float* __restrict__ b, __bf16* __restrict__ Y)
{
  __shared__ float red[256];
  const long row = blockIdx.x;
  const int  t   = threadIdx.x;
  const float* x = X + row * VIT_DIM;
  float v0 = x[t], v1 = x[t + 256], v2 = x[t + 512];
  red[t] = v0 + v1 + v2;
  __syncthreads();
  for (int o = 128; o > 0; o >>= 1) { if (t < o) red[t] += red[t + o]; __syncthreads(); }
  float mean = red[0] * (1.0f / VIT_DIM);
  __syncthreads();
  float d0 = v0 - mean, d1 = v1 - mean, d2 = v2 - mean;
  red[t] = d0*d0 + d1*d1 + d2*d2;
  __syncthreads();
  for (int o = 128; o > 0; o >>= 1) { if (t < o) red[t] += red[t + o]; __syncthreads(); }
  float rstd = rsqrtf(red[0] * (1.0f / VIT_DIM) + 1e-5f);
  __bf16* y = Y + row * VIT_DIM;
  y[t]       = (__bf16)(d0 * rstd * g[t]       + b[t]);
  y[t + 256] = (__bf16)(d1 * rstd * g[t + 256] + b[t + 256]);
  y[t + 512] = (__bf16)(d2 * rstd * g[t + 512] + b[t + 512]);
}

// ---------------------------------------------------------------------------
// TDM helper: issue one 2-D tile DMA (global -> LDS) per CDNA5 D# spec (§8).
//   g0: count=1 | lds_addr | global_addr(57b) | type=2
//   g1: data_size=2B, pad cfg, tensor dims / tile dims / dim0 stride
// ---------------------------------------------------------------------------
#if GEMM_TDM
__device__ __forceinline__ void tdm_load_2d(
    unsigned long long gaddr, unsigned lds_addr, unsigned w0,
    unsigned tensor_d0, unsigned tensor_d1,
    unsigned tile_d0, unsigned tile_d1, unsigned long long stride0)
{
  uint4v g0;
  g0.x = 1u;                                            // count=1, user desc
  g0.y = lds_addr;                                      // LDS byte address
  g0.z = (unsigned)gaddr;                               // global_addr[31:0]
  g0.w = (unsigned)((gaddr >> 32) & 0x01FFFFFFull) | (2u << 30);  // [56:32] | type=2
  int8v g1;
  g1[0] = (int)w0;                                      // data_size | pad cfg
  g1[1] = (int)((tensor_d0 & 0xFFFFu) << 16);           // tensor_dim0[15:0]
  g1[2] = (int)(((tensor_d0 >> 16) & 0xFFFFu) | ((tensor_d1 & 0xFFFFu) << 16));
  g1[3] = (int)(((tensor_d1 >> 16) & 0xFFFFu) | ((tile_d0 & 0xFFFFu) << 16));
  g1[4] = (int)(tile_d1 & 0xFFFFu);                     // tile_dim1 (tile_dim2=0)
  g1[5] = (int)(stride0 & 0xFFFFFFFFull);               // dim0 stride lo
  g1[6] = (int)((stride0 >> 32) & 0xFFFFull);           // dim0 stride hi (dim1 stride=0)
  g1[7] = 0;
  int4v z4 = {0, 0, 0, 0};
#if defined(__clang_major__) && (__clang_major__ >= 23)
  int8v z8 = {0, 0, 0, 0, 0, 0, 0, 0};
  __builtin_amdgcn_tensor_load_to_lds(g0, g1, z4, z4, z8, 0);
#else
  __builtin_amdgcn_tensor_load_to_lds(g0, g1, z4, z4, 0);
#endif
}
#endif

// ---------------------------------------------------------------------------
// WMMA GEMM: C[M,N] = act(A[M,K](bf16) @ W[K,N](bf16) + bias)
// Block tile 128x256x32; 256 threads = 8 waves (2 row x 4 col), wave tile 64x64
// => 16 v_wmma per k-step per wave.  TDM double-buffered LDS staging when
// available (wave 0 drives the DMA, s_wait_tensorcnt + barrier sync), else
// synchronous vector loads.
// MODE 0: store bf16   MODE 1: GELU(exact)+store bf16   MODE 2: fp32 C += result
// ---------------------------------------------------------------------------
#define GM 128
#define GN 256
#define GK 32
#define AS_LD 40    // 32 + 8 pad  (pad 4 dwords every 16 dwords)
#define BS_LD 264   // 256 + 8 pad (pad 4 dwords every 128 dwords)
#define ABUF_ELEMS (GM * AS_LD)
#define BBUF_ELEMS (GK * BS_LD)

template<int MODE>
__global__ __launch_bounds__(256) void gemm_wmma_bf16(
    const __bf16* __restrict__ A, const __bf16* __restrict__ W,
    const float* __restrict__ bias, void* __restrict__ Cout,
    int N, int K)
{
#if GEMM_TDM
  __shared__ __align__(16) __bf16 As[2 * ABUF_ELEMS];
  __shared__ __align__(16) __bf16 Bs[2 * BBUF_ELEMS];
#else
  __shared__ __align__(16) __bf16 As[ABUF_ELEMS];
  __shared__ __align__(16) __bf16 Bs[BBUF_ELEMS];
#endif

  const int tid  = threadIdx.x;
  const int lane = tid & 31;
  const int wave = tid >> 5;        // 0..7
  const int wm   = wave >> 2;       // 0..1  (row group of 64)
  const int wn   = wave & 3;        // 0..3  (col group of 64)
  const int half = lane >> 4;       // 0/1
  const int l16  = lane & 15;

  const long m0 = (long)blockIdx.x * GM;
  const int  n0 = blockIdx.y * GN;

  float8 acc[4][4];
#pragma unroll
  for (int i = 0; i < 4; ++i)
#pragma unroll
    for (int j = 0; j < 4; ++j) acc[i][j] = zero8();

  const int nSteps = K / GK;

#if GEMM_TDM
  // pad cfg: data_size=2B (bit16), pad_enable (bit20), interval code (22:24),
  // amount code (25:31).  A row: 16 dwords -> code 3; B row: 128 dwords -> code 6;
  // pad amount 4 dwords -> code 3  (reproduces AS_LD=40 / BS_LD=264 strides).
  const unsigned W0A = (1u << 16) | (1u << 20) | (3u << 22) | (3u << 25);
  const unsigned W0B = (1u << 16) | (1u << 20) | (6u << 22) | (3u << 25);
  const unsigned ldsA = (unsigned)(unsigned long long)(void*)As;
  const unsigned ldsB = (unsigned)(unsigned long long)(void*)Bs;

  auto issueTile = [&](int step, int buf) {
    const long k0 = (long)step * GK;
    tdm_load_2d((unsigned long long)(unsigned long long)(const void*)(A + m0 * (long)K + k0),
                ldsA + (unsigned)buf * (ABUF_ELEMS * 2u), W0A,
                (unsigned)(K - k0), (unsigned)(VIT_M - m0),
                GK, GM, (unsigned long long)K);
    tdm_load_2d((unsigned long long)(unsigned long long)(const void*)(W + k0 * (long)N + n0),
                ldsB + (unsigned)buf * (BBUF_ELEMS * 2u), W0B,
                (unsigned)(N - n0), (unsigned)(K - k0),
                GN, GK, (unsigned long long)N);
  };

  if (wave == 0) issueTile(0, 0);

  for (int step = 0; step < nSteps; ++step) {
    const int cur = step & 1;
    if (wave == 0) __builtin_amdgcn_s_wait_tensorcnt(0);
    __syncthreads();                                   // tile `step` visible to all
    if (wave == 0 && step + 1 < nSteps) issueTile(step + 1, cur ^ 1);  // overlap DMA

    const __bf16* as = As + cur * ABUF_ELEMS;
    const __bf16* bs = Bs + cur * BBUF_ELEMS;
#else
  const int arow = tid >> 1;            // 0..127
  const int acol = (tid & 1) * 16;      // 0/16
  const int brow = tid >> 3;            // 0..31
  const int bseg = (tid & 7) * 32;      // 0..224

  for (int step = 0; step < nSteps; ++step) {
    const int k0 = step * GK;
    { // stage A 128x32
      const __bf16* src = A + (m0 + arow) * (long)K + k0 + acol;
      *(bhalf8*)&As[arow * AS_LD + acol]     = *(const bhalf8*)(src);
      *(bhalf8*)&As[arow * AS_LD + acol + 8] = *(const bhalf8*)(src + 8);
    }
    { // stage B 32x256
      const __bf16* src = W + (long)(k0 + brow) * N + n0 + bseg;
#pragma unroll
      for (int j = 0; j < 4; ++j)
        *(bhalf8*)&Bs[brow * BS_LD + bseg + j * 8] = *(const bhalf8*)(src + j * 8);
    }
    __syncthreads();
    const __bf16* as = As;
    const __bf16* bs = Bs;
#endif

    bhalf16 afrag[4], bfrag[4];
#pragma unroll
    for (int mf = 0; mf < 4; ++mf) {
      const int r = wm * 64 + mf * 16 + l16;          // A lane holds row lane%16
      bhalf8 lo = *(const bhalf8*)&as[r * AS_LD + half * 8];
      bhalf8 hi = *(const bhalf8*)&as[r * AS_LD + 16 + half * 8];
      afrag[mf] = join8(lo, hi);
    }
#pragma unroll
    for (int nf = 0; nf < 4; ++nf) {
      const int c = wn * 64 + nf * 16 + l16;          // B lane holds column lane%16
#pragma unroll
      for (int s = 0; s < 16; ++s) {
        const int kk = (s < 8) ? (half * 8 + s) : (16 + half * 8 + (s - 8));
        bfrag[nf][s] = bs[kk * BS_LD + c];
      }
    }
#pragma unroll
    for (int mf = 0; mf < 4; ++mf)
#pragma unroll
      for (int nf = 0; nf < 4; ++nf)
        acc[mf][nf] = __builtin_amdgcn_wmma_f32_16x16x32_bf16(
            false, afrag[mf], false, bfrag[nf], (short)0, acc[mf][nf], false, false);
    __syncthreads();
  }

  // Epilogue: C/D layout -> element v: row = v + 8*half, col = l16
#pragma unroll
  for (int mf = 0; mf < 4; ++mf) {
#pragma unroll
    for (int nf = 0; nf < 4; ++nf) {
      const int col = n0 + wn * 64 + nf * 16 + l16;
      const float bv = bias[col];
#pragma unroll
      for (int v = 0; v < 8; ++v) {
        const long row = m0 + wm * 64 + mf * 16 + v + half * 8;
        float val = acc[mf][nf][v] + bv;
        if (MODE == 0) {
          ((__bf16*)Cout)[row * N + col] = (__bf16)val;
        } else if (MODE == 1) {
          float gl = 0.5f * val * (1.0f + erff(val * 0.70710678118f));
          ((__bf16*)Cout)[row * N + col] = (__bf16)gl;
        } else {
          float* X = (float*)Cout;
          X[row * N + col] += val;              // fused residual add
        }
      }
    }
  }
}

// ---------------------------------------------------------------------------
// Fused attention: one block per (batch*head, 16-query tile).
// QKV buffer is bf16 [12288, 2304]: q at col h*64, k at 768+h*64, v at 1536+h*64.
// Tiles 4..7 (tokens 64..127) attend only to tokens 64..127 (inner-token path);
// all other tiles attend to all 384 keys. out_token == query_token.
// ---------------------------------------------------------------------------
#define QS_LD 72     // 64 + 8
#define KT_LD 392    // 384 + 8
#define SC_LD 388    // 384 + 4
#define PS_LD 392

__global__ __launch_bounds__(128) void attention_wmma(
    const __bf16* __restrict__ qkv, __bf16* __restrict__ attn)
{
  const int bh = blockIdx.x;
  const int b  = bh / VIT_HEADS, h = bh % VIT_HEADS;
  const int qt = blockIdx.y;                 // 0..23
  const int qtok0 = qt * 16;
  const bool inner = (qt >= 4 && qt < 8);
  const int keyBase = inner ? 64 : 0;
  const int nKeys   = inner ? 64 : VIT_NTOK;

  __shared__ __align__(16) __bf16 Qs[16 * QS_LD];
  __shared__ __align__(16) __bf16 Kt[64 * KT_LD];   // K transposed: [d][key]
  __shared__ __align__(16) float  Sc[16 * SC_LD];
  __shared__ __align__(16) __bf16 Ps[16 * PS_LD];
  __shared__ float sred[128];

  const int tid  = threadIdx.x;
  const int lane = tid & 31;
  const int wave = tid >> 5;    // 0..3
  const int half = lane >> 4;
  const int l16  = lane & 15;
  const long rowBase = (long)b * VIT_NTOK;

  // stage Q tile [16 x 64]
  {
    const int r = tid >> 3, seg = tid & 7;
    const __bf16* src = qkv + (rowBase + qtok0 + r) * 2304 + h * 64 + seg * 8;
    *(bhalf8*)&Qs[r * QS_LD + seg * 8] = *(const bhalf8*)src;
  }
  // stage K^T: vector-load 8 head dims of one key, scatter down a column
  for (int i = tid; i < nKeys * 8; i += 128) {
    const int key = i >> 3, seg = i & 7;
    bhalf8 vv = *(const bhalf8*)(qkv + (rowBase + keyBase + key) * 2304 + 768 + h * 64 + seg * 8);
#pragma unroll
    for (int j = 0; j < 8; ++j) Kt[(seg * 8 + j) * KT_LD + key] = vv[j];
  }
  __syncthreads();

  // scores S[16, nKeys] = (Q @ K^T) * 1/sqrt(64)
  const int nKT = nKeys >> 4;
  for (int kt = wave; kt < nKT; kt += 4) {
    float8 acc = zero8();
#pragma unroll
    for (int d0 = 0; d0 < 64; d0 += 32) {
      bhalf8 alo = *(const bhalf8*)&Qs[l16 * QS_LD + d0 + half * 8];
      bhalf8 ahi = *(const bhalf8*)&Qs[l16 * QS_LD + d0 + 16 + half * 8];
      bhalf16 a = join8(alo, ahi);
      bhalf16 bb;
#pragma unroll
      for (int s = 0; s < 16; ++s) {
        const int kk = (s < 8) ? (half * 8 + s) : (16 + half * 8 + (s - 8));
        bb[s] = Kt[(d0 + kk) * KT_LD + kt * 16 + l16];
      }
      acc = __builtin_amdgcn_wmma_f32_16x16x32_bf16(false, a, false, bb, (short)0, acc, false, false);
    }
#pragma unroll
    for (int v = 0; v < 8; ++v)
      Sc[(v + half * 8) * SC_LD + kt * 16 + l16] = acc[v] * 0.125f;
  }
  __syncthreads();

  // softmax: 8 threads per query row
  {
    const int srow = tid >> 3;      // 0..15
    const int ssub = tid & 7;
    float mx = -1e30f;
    for (int k = ssub; k < nKeys; k += 8) mx = fmaxf(mx, Sc[srow * SC_LD + k]);
    sred[tid] = mx;
    __syncthreads();
    if (ssub < 4) sred[tid] = fmaxf(sred[tid], sred[tid + 4]);
    __syncthreads();
    if (ssub < 2) sred[tid] = fmaxf(sred[tid], sred[tid + 2]);
    __syncthreads();
    if (ssub == 0) sred[tid] = fmaxf(sred[tid], sred[tid + 1]);
    __syncthreads();
    mx = sred[srow * 8];
    float sum = 0.0f;
    for (int k = ssub; k < nKeys; k += 8) {
      float e = __expf(Sc[srow * SC_LD + k] - mx);
      Sc[srow * SC_LD + k] = e;
      sum += e;
    }
    __syncthreads();
    sred[tid] = sum;
    __syncthreads();
    if (ssub < 4) sred[tid] += sred[tid + 4];
    __syncthreads();
    if (ssub < 2) sred[tid] += sred[tid + 2];
    __syncthreads();
    if (ssub == 0) sred[tid] += sred[tid + 1];
    __syncthreads();
    const float inv = 1.0f / sred[srow * 8];
    for (int k = ssub; k < nKeys; k += 8)
      Ps[srow * PS_LD + k] = (__bf16)(Sc[srow * SC_LD + k] * inv);
  }
  __syncthreads();

  // O[16, 64] = P @ V ; wave handles d-tile = wave*16; V read straight from global
  {
    float8 acc = zero8();
    const int dcol = h * 64 + wave * 16;
    for (int kc = 0; kc < nKeys; kc += 32) {
      bhalf8 alo = *(const bhalf8*)&Ps[l16 * PS_LD + kc + half * 8];
      bhalf8 ahi = *(const bhalf8*)&Ps[l16 * PS_LD + kc + 16 + half * 8];
      bhalf16 a = join8(alo, ahi);
      bhalf16 bb;
#pragma unroll
      for (int s = 0; s < 16; ++s) {
        const int kk = (s < 8) ? (half * 8 + s) : (16 + half * 8 + (s - 8));
        bb[s] = qkv[(rowBase + keyBase + kc + kk) * 2304 + 1536 + dcol + l16];
      }
      acc = __builtin_amdgcn_wmma_f32_16x16x32_bf16(false, a, false, bb, (short)0, acc, false, false);
    }
#pragma unroll
    for (int v = 0; v < 8; ++v)
      attn[(rowBase + qtok0 + v + half * 8) * VIT_DIM + dcol + l16] = (__bf16)acc[v];
  }
}

// ---------------------------------------------------------------------------
// Host orchestration
// ---------------------------------------------------------------------------
extern "C" void kernel_launch(void* const* d_in, const int* in_sizes, int n_in,
                              void* d_out, int out_size, void* d_ws, size_t ws_size,
                              hipStream_t stream) {
  (void)in_sizes; (void)n_in; (void)out_size; (void)ws_size;

  const float* in_x      = (const float*)d_in[0];
  const float* in_qkv_w  = (const float*)d_in[1];
  const float* in_qkv_b  = (const float*)d_in[2];
  const float* in_proj_w = (const float*)d_in[3];
  const float* in_proj_b = (const float*)d_in[4];
  const float* in_ln1_g  = (const float*)d_in[5];
  const float* in_ln1_b  = (const float*)d_in[6];
  const float* in_ln2_g  = (const float*)d_in[7];
  const float* in_ln2_b  = (const float*)d_in[8];
  const float* in_fc1_w  = (const float*)d_in[9];
  const float* in_fc1_b  = (const float*)d_in[10];
  const float* in_fc2_w  = (const float*)d_in[11];
  const float* in_fc2_b  = (const float*)d_in[12];

  // workspace layout (bytes)
  char* ws = (char*)d_ws;
  const size_t SZ_X    = (size_t)VIT_M * VIT_DIM * 4;          // fp32 activations
  const size_t SZ_XN   = (size_t)VIT_M * VIT_DIM * 2;          // bf16 LN output
  const size_t SZ_QKV  = (size_t)VIT_M * 3 * VIT_DIM * 2;      // bf16 qkv
  const size_t SZ_ATTN = (size_t)VIT_M * VIT_DIM * 2;          // bf16 attn out
  const size_t SZ_H    = (size_t)VIT_M * VIT_MLP * 2;          // bf16 mlp hidden
  const size_t SZ_WQKV = (size_t)VIT_DEPTH * VIT_DIM * 3 * VIT_DIM * 2;
  const size_t SZ_WPRJ = (size_t)VIT_DEPTH * VIT_DIM * VIT_DIM * 2;
  const size_t SZ_WFC1 = (size_t)VIT_DEPTH * VIT_DIM * VIT_MLP * 2;

  float*  X     = (float*) (ws);
  __bf16* XN    = (__bf16*)(ws + SZ_X);
  __bf16* QKV   = (__bf16*)(ws + SZ_X + SZ_XN);
  __bf16* ATTN  = (__bf16*)(ws + SZ_X + SZ_XN + SZ_QKV);
  __bf16* Hbuf  = (__bf16*)(ws + SZ_X + SZ_XN + SZ_QKV + SZ_ATTN);
  __bf16* WQKV  = (__bf16*)(ws + SZ_X + SZ_XN + SZ_QKV + SZ_ATTN + SZ_H);
  __bf16* WPROJ = (__bf16*)((char*)WQKV  + SZ_WQKV);
  __bf16* WFC1  = (__bf16*)((char*)WPROJ + SZ_WPRJ);
  __bf16* WFC2  = (__bf16*)((char*)WFC1  + SZ_WFC1);

  // one-time (per launch) weight conversion fp32 -> bf16
  {
    const long nq = (long)VIT_DEPTH * VIT_DIM * 3 * VIT_DIM;
    const long np = (long)VIT_DEPTH * VIT_DIM * VIT_DIM;
    const long n1 = (long)VIT_DEPTH * VIT_DIM * VIT_MLP;
    const long n2 = (long)VIT_DEPTH * VIT_MLP * VIT_DIM;
    cvt_f32_bf16<<<2048, 256, 0, stream>>>(in_qkv_w,  WQKV,  nq);
    cvt_f32_bf16<<<2048, 256, 0, stream>>>(in_proj_w, WPROJ, np);
    cvt_f32_bf16<<<2048, 256, 0, stream>>>(in_fc1_w,  WFC1,  n1);
    cvt_f32_bf16<<<2048, 256, 0, stream>>>(in_fc2_w,  WFC2,  n2);
  }

  // x working copy (fp32, updated in place by residual GEMMs)
  hipMemcpyAsync(X, in_x, SZ_X, hipMemcpyDeviceToDevice, stream);

  const dim3 blk256(256), blk128(128);
  const dim3 gridLN(VIT_M);
  const dim3 gridQKV (VIT_M / GM, (3 * VIT_DIM) / GN);  // 96 x 9
  const dim3 gridPROJ(VIT_M / GM, VIT_DIM / GN);        // 96 x 3
  const dim3 gridFC1 (VIT_M / GM, VIT_MLP / GN);        // 96 x 12
  const dim3 gridATT (VIT_B * VIT_HEADS, VIT_NTOK / 16);// 384 x 24

  for (int l = 0; l < VIT_DEPTH; ++l) {
    // 1) xn = LN1(x)
    layernorm_bf16<<<gridLN, blk256, 0, stream>>>(
        X, in_ln1_g + (long)l * VIT_DIM, in_ln1_b + (long)l * VIT_DIM, XN);
    // 2) qkv = xn @ Wqkv + b  (bf16 out)
    gemm_wmma_bf16<0><<<gridQKV, blk256, 0, stream>>>(
        XN, WQKV + (long)l * VIT_DIM * 3 * VIT_DIM,
        in_qkv_b + (long)l * 3 * VIT_DIM, QKV, 3 * VIT_DIM, VIT_DIM);
    // 3) fused dual attention
    attention_wmma<<<gridATT, blk128, 0, stream>>>(QKV, ATTN);
    // 4) x += attn @ Wproj + b
    gemm_wmma_bf16<2><<<gridPROJ, blk256, 0, stream>>>(
        ATTN, WPROJ + (long)l * VIT_DIM * VIT_DIM,
        in_proj_b + (long)l * VIT_DIM, X, VIT_DIM, VIT_DIM);
    // 5) xn = LN2(x)
    layernorm_bf16<<<gridLN, blk256, 0, stream>>>(
        X, in_ln2_g + (long)l * VIT_DIM, in_ln2_b + (long)l * VIT_DIM, XN);
    // 6) h = gelu(xn @ Wfc1 + b)  (bf16 out)
    gemm_wmma_bf16<1><<<gridFC1, blk256, 0, stream>>>(
        XN, WFC1 + (long)l * VIT_DIM * VIT_MLP,
        in_fc1_b + (long)l * VIT_MLP, Hbuf, VIT_MLP, VIT_DIM);
    // 7) x += h @ Wfc2 + b
    gemm_wmma_bf16<2><<<gridPROJ, blk256, 0, stream>>>(
        Hbuf, WFC2 + (long)l * VIT_MLP * VIT_DIM,
        in_fc2_b + (long)l * VIT_DIM, X, VIT_DIM, VIT_MLP);
  }

  hipMemcpyAsync(d_out, X, SZ_X, hipMemcpyDeviceToDevice, stream);
}
